// FineGrainedPipelinedMoELayer_34815004901535
// MI455X (gfx1250) — compile-verified, matrix operand order checked by reading, not run on hardware
//
#include <hip/hip_runtime.h>
#include <hip/hip_bf16.h>

// ---------------- problem constants (from reference) ----------------
#define TK   4096   // B*S tokens
#define DD   1024   // model dim
#define DHH  4096   // expert hidden
#define DOO  1024   // out dim
#define EE   8      // experts

// ---------------- tiling ----------------
#define BM   64     // M tile per block (tokens)
#define BN   128    // N tile per block (8 waves * 16)
#define KC   32     // K step (bf16 wmma K)
#define KCP  40     // padded K stride in LDS (elements) = 64B row + 16B pad

typedef __attribute__((ext_vector_type(16))) __bf16 v16bf;
typedef __attribute__((ext_vector_type(8)))  __bf16 v8bf;
typedef __attribute__((ext_vector_type(8)))  float  v8f;
typedef __attribute__((ext_vector_type(4)))  unsigned int u32x4;
typedef __attribute__((ext_vector_type(4)))  int i32x4;
typedef __attribute__((ext_vector_type(8)))  int i32x8;

#if __has_builtin(__builtin_amdgcn_tensor_load_to_lds)
#define HAVE_TDM 1
#endif

// ---- s_wait_tensorcnt helper ----
__device__ __forceinline__ void wait_tensor2() {
#if __has_builtin(__builtin_amdgcn_s_wait_tensorcnt)
    __builtin_amdgcn_s_wait_tensorcnt(2);
#else
    asm volatile("s_wait_tensorcnt 0x2" ::: "memory");
#endif
}
__device__ __forceinline__ void wait_tensor0() {
#if __has_builtin(__builtin_amdgcn_s_wait_tensorcnt)
    __builtin_amdgcn_s_wait_tensorcnt(0);
#else
    asm volatile("s_wait_tensorcnt 0x0" ::: "memory");
#endif
}

// ---- TDM: DMA a 2D bf16 tile (rows x 32) from global into LDS with +16B row pad ----
// D# per ISA 8.3/8.4: g0 = {count=1, lds_addr, global_addr(57b), type=2}
// g1: data_size=2B, pad_enable, pad_interval code 3 (every 64B), pad_amount code 3 (+4 dwords)
//     tensor_dim == tile_dim (no clipping), tensor_dim0_stride = row stride (elements)
__device__ __forceinline__ void tdm_load_tile(unsigned ldsOff, const __bf16* gsrc,
                                              int rows, int strideElems) {
#ifdef HAVE_TDM
    const unsigned long long ga = (unsigned long long)(uintptr_t)gsrc;
    u32x4 g0;
    g0[0] = 1u;                                                   // count = 1 descriptor
    g0[1] = ldsOff;                                               // LDS byte offset
    g0[2] = (unsigned)ga;                                         // global addr lo
    g0[3] = ((unsigned)(ga >> 32) & 0x01FFFFFFu) | (2u << 30);    // addr hi | type=2
    i32x8 g1;
    g1[0] = (1 << 16) | (1 << 20) | (3 << 22) | (3 << 25);        // 2B elems, pad 64B->+16B
    g1[1] = (KC & 0xFFFF) << 16;                                  // tensor_dim0 = 32
    g1[2] = (rows & 0xFFFF) << 16;                                // tensor_dim1 = rows
    g1[3] = (KC & 0xFFFF) << 16;                                  // tile_dim0 = 32
    g1[4] = (rows & 0xFFFF);                                      // tile_dim1 = rows
    g1[5] = strideElems;                                          // dim0 stride (elements)
    g1[6] = 0;
    g1[7] = 0;
    i32x4 z4 = {0, 0, 0, 0};
#if defined(__clang_major__) && __clang_major__ >= 23
    i32x8 z8 = {0, 0, 0, 0, 0, 0, 0, 0};
    __builtin_amdgcn_tensor_load_to_lds(g0, g1, z4, z4, z8, 0);
#else
    __builtin_amdgcn_tensor_load_to_lds(g0, g1, z4, z4, 0);
#endif
#endif
}

__device__ __forceinline__ unsigned lds_off(const void* p) {
    return (unsigned)(uintptr_t)p;   // low 32 bits of flat addr = LDS byte offset
}

// ---- fallback staging copy (no TDM): rows x 32 bf16, padded KCP stride ----
__device__ __forceinline__ void copy_tile(__bf16* dst, const __bf16* src,
                                          int rows, int strideElems, int tid) {
    for (int c = tid; c < rows * 4; c += 256) {
        const int r = c >> 2, ks = (c & 3) * 8;
        *(v8bf*)&dst[r * KCP + ks] = *(const v8bf*)&src[(size_t)r * strideElems + ks];
    }
}

// Build a 16-element bf16 WMMA fragment per ISA 16-bit A/B layout:
//   elements 0..7  <- K = g*8 + 0..7 ; elements 8..15 <- K = 16 + g*8 + 0..7  (g = lane>>4)
__device__ __forceinline__ v16bf frag32(const __bf16* base, int g) {
    const v8bf lo = *(const v8bf*)(base + g * 8);
    const v8bf hi = *(const v8bf*)(base + 16 + g * 8);
    return __builtin_shufflevector(lo, hi, 0, 1, 2, 3, 4, 5, 6, 7,
                                           8, 9, 10, 11, 12, 13, 14, 15);
}

// ---------------- 1) gating: argmax(x @ Wg + bg) ----------------
__global__ __launch_bounds__(256) void moe_gate(const float* __restrict__ x,
                                                const float* __restrict__ Wg,
                                                const float* __restrict__ bg,
                                                int* __restrict__ sel) {
    const int t    = blockIdx.x * 8 + (threadIdx.x >> 5);
    const int lane = threadIdx.x & 31;
    const float* xr = x + (size_t)t * DD;
    float best = -3.0e38f;
    int   bi   = 0;
    for (int e = 0; e < EE; ++e) {
        float s = 0.f;
        for (int d = lane; d < DD; d += 32)
            s += xr[d] * Wg[d * EE + e];
#pragma unroll
        for (int off = 16; off; off >>= 1) s += __shfl_xor(s, off);
        s += bg[e];
        if (s > best) { best = s; bi = e; }   // first-max tie break = argmax
    }
    if (lane == 0) sel[t] = bi;
}

// ---------------- 2) routing ----------------
__global__ __launch_bounds__(256) void moe_route(const int* __restrict__ sel,
                                                 int* __restrict__ counts,
                                                 int* __restrict__ bucket) {
    const int t = blockIdx.x * 256 + threadIdx.x;
    if (t < TK) {
        const int e = sel[t];
        const int p = atomicAdd(&counts[e], 1);
        bucket[e * TK + p] = t;
    }
}

__global__ void moe_offsets(const int* __restrict__ counts, int* __restrict__ offs) {
    if (threadIdx.x == 0) {
        int a = 0;
        for (int e = 0; e < EE; ++e) { offs[e] = a; a += counts[e]; }
    }
}

__global__ __launch_bounds__(256) void moe_buildrows(const int* __restrict__ counts,
                                                     const int* __restrict__ offs,
                                                     const int* __restrict__ bucket,
                                                     int* __restrict__ rowtok) {
    const int e = blockIdx.x, cnt = counts[e], o = offs[e];
    for (int i = threadIdx.x; i < cnt; i += 256) rowtok[o + i] = bucket[e * TK + i];
}

// ---------------- 3) gather x rows into slot order, fp32 -> bf16 ----------------
__global__ __launch_bounds__(256) void moe_gather(const float* __restrict__ x,
                                                  const int* __restrict__ rowtok,
                                                  __bf16* __restrict__ xg) {
    const int r   = blockIdx.x;
    const int tok = rowtok[r];
    const int d   = threadIdx.x * 4;
    const float4 v = *(const float4*)(x + (size_t)tok * DD + d);
    __bf16* dst = xg + (size_t)r * DD + d;
    dst[0] = (__bf16)v.x; dst[1] = (__bf16)v.y;
    dst[2] = (__bf16)v.z; dst[3] = (__bf16)v.w;
}

// ---------------- 4) weight convert + transpose: W[e][K][N] fp32 -> Wt[e][N][K] bf16 ----------------
__global__ __launch_bounds__(256) void conv_transpose(const float* __restrict__ W,
                                                      __bf16* __restrict__ Wt,
                                                      int K, int N) {
    const size_t eOfs = (size_t)blockIdx.z * K * N;
    const int n0 = blockIdx.x * 32, k0 = blockIdx.y * 32;
    __shared__ __bf16 tile[32][33];
    const int tx = threadIdx.x & 31, ty = threadIdx.x >> 5;
#pragma unroll
    for (int j = 0; j < 4; ++j) {
        const int kk = ty + j * 8;
        tile[kk][tx] = (__bf16)W[eOfs + (size_t)(k0 + kk) * N + n0 + tx];
    }
    __syncthreads();
#pragma unroll
    for (int j = 0; j < 4; ++j) {
        const int nn = ty + j * 8;
        Wt[eOfs + (size_t)(n0 + nn) * K + k0 + tx] = tile[tx][nn];
    }
}

// ---------------- 5) grouped GEMM1: h = gelu(xg @ W1t^T + b1) in slot space ----------------
__global__ __launch_bounds__(256) void moe_gemm1(const __bf16* __restrict__ xg,
                                                 const __bf16* __restrict__ W1t,
                                                 const float* __restrict__ b1,
                                                 const int* __restrict__ counts,
                                                 const int* __restrict__ offs,
                                                 __bf16* __restrict__ h) {
    const int e     = blockIdx.z;
    const int cnt   = counts[e];
    const int mBase = blockIdx.y * BM;
    if (mBase >= cnt) return;
    const int nBase = blockIdx.x * BN;
    const int roff  = __builtin_amdgcn_readfirstlane(offs[e]);

    const int tid  = threadIdx.x;
    const int lane = tid & 31;
    const int wave = tid >> 5;
    const int g    = (lane >> 4) & 1;
    const int ln   = lane & 15;

    __shared__ __bf16 As[2][BM * KCP];
    __shared__ __bf16 Bs[2][BN * KCP];

    const __bf16* Asrc = xg  + (size_t)(roff + mBase) * DD;
    const __bf16* Bsrc = W1t + ((size_t)e * DHH + nBase) * DD;

#ifdef HAVE_TDM
    if (wave == 0) {
        tdm_load_tile(lds_off(&As[0][0]), Asrc, BM, DD);
        tdm_load_tile(lds_off(&Bs[0][0]), Bsrc, BN, DD);
    }
#else
    copy_tile(As[0], Asrc, BM, DD, tid);
    copy_tile(Bs[0], Bsrc, BN, DD, tid);
#endif

    v8f acc[4] = {};
    const int nk = DD / KC;
    for (int i = 0; i < nk; ++i) {
        const int cur = i & 1, nxt = cur ^ 1;
        __syncthreads();                         // prev compute done -> nxt buffer free
#ifdef HAVE_TDM
        if (wave == 0) {
            if (i + 1 < nk) {
                tdm_load_tile(lds_off(&As[nxt][0]), Asrc + (i + 1) * KC, BM, DD);
                tdm_load_tile(lds_off(&Bs[nxt][0]), Bsrc + (i + 1) * KC, BN, DD);
                wait_tensor2();                  // in-order: tile i complete
            } else {
                wait_tensor0();
            }
        }
#else
        if (i + 1 < nk) {
            copy_tile(As[nxt], Asrc + (i + 1) * KC, BM, DD, tid);
            copy_tile(Bs[nxt], Bsrc + (i + 1) * KC, BN, DD, tid);
        }
#endif
        __syncthreads();                         // tile i visible to all waves
        const __bf16* Ab = As[cur];
        const __bf16* Bb = Bs[cur];
        // load all fragments first (distinct regs), then drain the WMMA pipe
        const v16bf bf = frag32(&Bb[(wave * 16 + ln) * KCP], g);
        v16bf af[4];
#pragma unroll
        for (int tm = 0; tm < 4; ++tm)
            af[tm] = frag32(&Ab[(tm * 16 + ln) * KCP], g);
#pragma unroll
        for (int tm = 0; tm < 4; ++tm)
            acc[tm] = __builtin_amdgcn_wmma_f32_16x16x32_bf16(
                false, af[tm], false, bf, (short)0, acc[tm], false, false);
    }

    const int   col  = nBase + wave * 16 + ln;
    const float bias = b1[(size_t)e * DHH + col];
#pragma unroll
    for (int tm = 0; tm < 4; ++tm) {
#pragma unroll
        for (int r = 0; r < 8; ++r) {
            const int s = mBase + tm * 16 + r + g * 8;
            if (s < cnt) {
                const float v  = acc[tm][r] + bias;
                const float ge = 0.5f * v * (1.f + erff(v * 0.70710678118654752f));
                h[(size_t)(roff + s) * DHH + col] = (__bf16)ge;
            }
        }
    }
}

// ---------------- 6) grouped GEMM2: out = h @ W2t^T + b2, scatter by token ----------------
__global__ __launch_bounds__(256) void moe_gemm2(const __bf16* __restrict__ h,
                                                 const __bf16* __restrict__ W2t,
                                                 const float* __restrict__ b2,
                                                 const int* __restrict__ counts,
                                                 const int* __restrict__ offs,
                                                 const int* __restrict__ rowtok,
                                                 float* __restrict__ out) {
    const int e     = blockIdx.z;
    const int cnt   = counts[e];
    const int mBase = blockIdx.y * BM;
    if (mBase >= cnt) return;
    const int nBase = blockIdx.x * BN;
    const int roff  = __builtin_amdgcn_readfirstlane(offs[e]);

    const int tid  = threadIdx.x;
    const int lane = tid & 31;
    const int wave = tid >> 5;
    const int g    = (lane >> 4) & 1;
    const int ln   = lane & 15;

    __shared__ __bf16 As[2][BM * KCP];
    __shared__ __bf16 Bs[2][BN * KCP];
    __shared__ int    tokBuf[BM];

    if (tid < BM) {
        const int s = mBase + tid;
        tokBuf[tid] = (s < cnt) ? rowtok[roff + s] : -1;
    }

    const __bf16* Asrc = h   + (size_t)(roff + mBase) * DHH;
    const __bf16* Bsrc = W2t + ((size_t)e * DOO + nBase) * DHH;

#ifdef HAVE_TDM
    if (wave == 0) {
        tdm_load_tile(lds_off(&As[0][0]), Asrc, BM, DHH);
        tdm_load_tile(lds_off(&Bs[0][0]), Bsrc, BN, DHH);
    }
#else
    copy_tile(As[0], Asrc, BM, DHH, tid);
    copy_tile(Bs[0], Bsrc, BN, DHH, tid);
#endif

    v8f acc[4] = {};
    const int nk = DHH / KC;
    for (int i = 0; i < nk; ++i) {
        const int cur = i & 1, nxt = cur ^ 1;
        __syncthreads();
#ifdef HAVE_TDM
        if (wave == 0) {
            if (i + 1 < nk) {
                tdm_load_tile(lds_off(&As[nxt][0]), Asrc + (i + 1) * KC, BM, DHH);
                tdm_load_tile(lds_off(&Bs[nxt][0]), Bsrc + (i + 1) * KC, BN, DHH);
                wait_tensor2();
            } else {
                wait_tensor0();
            }
        }
#else
        if (i + 1 < nk) {
            copy_tile(As[nxt], Asrc + (i + 1) * KC, BM, DHH, tid);
            copy_tile(Bs[nxt], Bsrc + (i + 1) * KC, BN, DHH, tid);
        }
#endif
        __syncthreads();
        const __bf16* Ab = As[cur];
        const __bf16* Bb = Bs[cur];
        const v16bf bf = frag32(&Bb[(wave * 16 + ln) * KCP], g);
        v16bf af[4];
#pragma unroll
        for (int tm = 0; tm < 4; ++tm)
            af[tm] = frag32(&Ab[(tm * 16 + ln) * KCP], g);
#pragma unroll
        for (int tm = 0; tm < 4; ++tm)
            acc[tm] = __builtin_amdgcn_wmma_f32_16x16x32_bf16(
                false, af[tm], false, bf, (short)0, acc[tm], false, false);
    }

    const int   col  = nBase + wave * 16 + ln;
    const float bias = b2[(size_t)e * DOO + col];
#pragma unroll
    for (int tm = 0; tm < 4; ++tm) {
#pragma unroll
        for (int r = 0; r < 8; ++r) {
            const int s = mBase + tm * 16 + r + g * 8;
            if (s < cnt) {
                const int tok = tokBuf[tm * 16 + r + g * 8];
                out[(size_t)tok * DOO + col] = acc[tm][r] + bias;
            }
        }
    }
}

// ---------------- host-side launch ----------------
extern "C" void kernel_launch(void* const* d_in, const int* in_sizes, int n_in,
                              void* d_out, int out_size, void* d_ws, size_t ws_size,
                              hipStream_t stream) {
    const float* x  = (const float*)d_in[0];
    const float* Wg = (const float*)d_in[1];
    const float* bg = (const float*)d_in[2];
    const float* W1 = (const float*)d_in[3];
    const float* b1 = (const float*)d_in[4];
    const float* W2 = (const float*)d_in[5];
    const float* b2 = (const float*)d_in[6];
    float* out = (float*)d_out;

    // workspace layout (256B aligned chunks)
    char*  base = (char*)d_ws;
    size_t off  = 0;
    auto alloc = [&](size_t bytes) -> void* {
        void* r = base + off;
        off = (off + bytes + 255) & ~(size_t)255;
        return r;
    };
    int*    sel    = (int*)alloc(TK * sizeof(int));
    int*    counts = (int*)alloc(64 * sizeof(int));
    int*    offs   = (int*)alloc(16 * sizeof(int));
    int*    bucket = (int*)alloc((size_t)EE * TK * sizeof(int));
    int*    rowtok = (int*)alloc(TK * sizeof(int));
    __bf16* xg     = (__bf16*)alloc(((size_t)TK + BM) * DD * sizeof(__bf16));   // + tile slack
    __bf16* h      = (__bf16*)alloc(((size_t)TK + BM) * DHH * sizeof(__bf16));  // + tile slack
    __bf16* W1t    = (__bf16*)alloc((size_t)EE * DHH * DD * sizeof(__bf16));
    __bf16* W2t    = (__bf16*)alloc((size_t)EE * DOO * DHH * sizeof(__bf16));

    // routing
    moe_gate<<<TK / 8, 256, 0, stream>>>(x, Wg, bg, sel);
    hipMemsetAsync(counts, 0, 64 * sizeof(int), stream);
    moe_route<<<TK / 256, 256, 0, stream>>>(sel, counts, bucket);
    moe_offsets<<<1, 32, 0, stream>>>(counts, offs);
    moe_buildrows<<<EE, 256, 0, stream>>>(counts, offs, bucket, rowtok);

    // operand prep (bf16, slot-packed / transposed)
    moe_gather<<<TK, 256, 0, stream>>>(x, rowtok, xg);
    {
        dim3 gw1(DHH / 32, DD / 32, EE);
        conv_transpose<<<gw1, 256, 0, stream>>>(W1, W1t, DD, DHH);
        dim3 gw2(DOO / 32, DHH / 32, EE);
        conv_transpose<<<gw2, 256, 0, stream>>>(W2, W2t, DHH, DOO);
    }

    // grouped GEMMs (TDM-fed, double buffered)
    dim3 g1(DHH / BN, TK / BM, EE);
    moe_gemm1<<<g1, 256, 0, stream>>>(xg, W1t, b1, counts, offs, h);
    dim3 g2(DOO / BN, TK / BM, EE);
    moe_gemm2<<<g2, 256, 0, stream>>>(h, W2t, b2, counts, offs, rowtok, out);
}
